// LabelEmbedder_deeplesion_79980880986306
// MI455X (gfx1250) — matrix operand order repaired and checked by reading, not compiled, not run on hardware
//
#include <hip/hip_runtime.h>

typedef __attribute__((ext_vector_type(16))) _Float16 v16h;
typedef __attribute__((ext_vector_type(8)))  _Float16 v8h;
typedef __attribute__((ext_vector_type(8)))  float    v8f;
typedef int v4i __attribute__((vector_size(16)));   // matches builtin's pointee type

#define OUT_RELU_F16 0
#define OUT_H        1
#define OUT_ADD_H    2

#if __has_builtin(__builtin_amdgcn_global_load_async_to_lds_b128)
#define HAVE_ASYNC_LDS 1
#else
#define HAVE_ASYNC_LDS 0
#endif

__device__ __forceinline__ void wait_asynccnt0() {
#if __has_builtin(__builtin_amdgcn_s_wait_asynccnt)
    __builtin_amdgcn_s_wait_asynccnt(0);
#else
    asm volatile("s_wait_asynccnt 0x0" ::: "memory");
#endif
}

// ---------------- weight prep: OIHW f32 -> [n][K] f16 with k = tap*CI + ci
template<int N, int K, int CI>
__global__ __launch_bounds__(256) void prep_weights(
    const float* __restrict__ src, _Float16* __restrict__ dst)
{
    int id = blockIdx.x * 256 + threadIdx.x;
    if (id >= N * K) return;
    int n = id / K;
    int k = id - n * K;
    int tap = k / CI;
    int ci  = k - tap * CI;
    dst[id] = (_Float16)src[(n * CI + ci) * (K / CI) + tap];
}

// ---------------- conv1: [32,1,128,128] -> NHWC f16 [32,64,64,64], 4x4 s2 p1, relu
__global__ __launch_bounds__(256) void conv1_kernel(
    const float* __restrict__ x, const float* __restrict__ w1,
    const float* __restrict__ b1, _Float16* __restrict__ act1)
{
    int idx = blockIdx.x * 256 + threadIdx.x;   // ((b*64+oy)*64+ox)*64+co
    int co = idx & 63;
    int p  = idx >> 6;
    int ox = p & 63; p >>= 6;
    int oy = p & 63;
    int b  = p >> 6;
    float acc = b1[co];
    const float* wp = w1 + co * 16;
    const float* xb = x + b * 16384;
#pragma unroll
    for (int ky = 0; ky < 4; ++ky) {
        int iy = 2 * oy - 1 + ky;
        bool vy = (iy >= 0) && (iy < 128);
#pragma unroll
        for (int kx = 0; kx < 4; ++kx) {
            int ix = 2 * ox - 1 + kx;
            bool v = vy && (ix >= 0) && (ix < 128);
            float xv = v ? xb[iy * 128 + ix] : 0.f;
            acc += wp[ky * 4 + kx] * xv;
        }
    }
    act1[idx] = (_Float16)(acc > 0.f ? acc : 0.f);
}

// ---------------- implicit-GEMM conv via WMMA f32_16x16x32_f16
// Block: 32 output rows (2 M-tiles), NOUT/16 waves (wave w = N-tile w).
// A staged per-tap into LDS via GLOBAL_LOAD_ASYNC_TO_LDS (ASYNCcnt), padding
// rows zero-filled with ds stores. B fragments loaded straight from the
// pre-transposed f16 weight image in global (L2-resident, two 16B loads/lane).
template<int CI, int KH, int KW, int STRIDE, int PAD, int IH, int IW, int NOUT, int MODE>
__global__ __launch_bounds__((NOUT / 16) * 32) void gemm_conv(
    const _Float16* __restrict__ Aact,   // NHWC activations
    const _Float16* __restrict__ Bprep,  // [NOUT][K] f16, k = tap*CI + ci
    const float*    __restrict__ bias,   // nullable
    _Float16*       __restrict__ outF16, // f16 output (ld = NOUT or 128)
    float*          __restrict__ hbuf)   // f32 h (ld=128) for OUT_H/OUT_ADD_H
{
    constexpr int NT   = (NOUT / 16) * 32;  // threads per block
    constexpr int KTOT = KH * KW * CI;
    constexpr int CI8  = CI / 8;            // uint4s per A row
    __shared__ __align__(64) _Float16 Alds[32 * CI];

    const int tid  = threadIdx.x;
    const int wave = tid >> 5;
    const int lane = tid & 31;
    const int nBase = wave * 16;
    const int mBase = blockIdx.x * 32;

    const bool hi = lane >= 16;
    const int  mr = lane & 15;

    // per-lane B fragment base (column nBase+mr, ISA 16-bit K layout)
    const _Float16* bcol = Bprep + (size_t)(nBase + mr) * KTOT + (hi ? 8 : 0);

    v8f acc0 = {}, acc1 = {};

    for (int tap = 0; tap < KH * KW; ++tap) {
        const int ky = tap / KW, kx = tap - ky * KW;

        __syncthreads();
        // cooperative A staging: 32 rows x CI channels for this tap
        for (int q = tid; q < 4 * CI; q += NT) {
            int row   = q / CI8;
            int cOff8 = q - row * CI8;      // uint4 index within row
            int m   = mBase + row;
            int b   = m >> 10;
            int rem = m & 1023;
            int oy  = rem >> 5;
            int ox  = rem & 31;
            int iy  = oy * STRIDE - PAD + ky;
            int ix  = ox * STRIDE - PAD + kx;
            bool valid = (iy >= 0) && (iy < IH) && (ix >= 0) && (ix < IW);
            const _Float16* gp = Aact +
                (size_t)((b * IH + iy) * IW + ix) * CI + cOff8 * 8;
            _Float16* lp = Alds + row * CI + cOff8 * 8;
#if HAVE_ASYNC_LDS
            if (valid) {
                __builtin_amdgcn_global_load_async_to_lds_b128(
                    (__attribute__((address_space(1))) v4i*)gp,
                    (__attribute__((address_space(3))) v4i*)lp, 0, 0);
            } else {
                uint4 z = {};
                *(uint4*)lp = z;            // zero padding row (DScnt)
            }
#else
            uint4 v = {};
            if (valid) v = *(const uint4*)gp;
            *(uint4*)lp = v;
#endif
        }
#if HAVE_ASYNC_LDS
        wait_asynccnt0();
#endif
        __syncthreads();

#pragma unroll
        for (int cc = 0; cc < CI / 32; ++cc) {
            // B fragment: f16 k 0-7 and 16-23 (+8 for hi lanes) from global
            const _Float16* bp = bcol + tap * CI + cc * 32;
            v8h b0 = *(const v8h*)bp;
            v8h b1 = *(const v8h*)(bp + 16);
            v16h bf = __builtin_shufflevector(b0, b1,
                0,1,2,3,4,5,6,7,8,9,10,11,12,13,14,15);

            // A fragments for the two M tiles (ISA 16-bit A layout)
            const _Float16* a0p = Alds + mr * CI + cc * 32 + (hi ? 8 : 0);
            const _Float16* a1p = a0p + 16 * CI;
            v8h a0lo = *(const v8h*)a0p, a0hi = *(const v8h*)(a0p + 16);
            v8h a1lo = *(const v8h*)a1p, a1hi = *(const v8h*)(a1p + 16);
            v16h a0 = __builtin_shufflevector(a0lo, a0hi,
                0,1,2,3,4,5,6,7,8,9,10,11,12,13,14,15);
            v16h a1 = __builtin_shufflevector(a1lo, a1hi,
                0,1,2,3,4,5,6,7,8,9,10,11,12,13,14,15);

            acc0 = __builtin_amdgcn_wmma_f32_16x16x32_f16(
                false, a0, false, bf, (short)0, acc0, false, false);
            acc1 = __builtin_amdgcn_wmma_f32_16x16x32_f16(
                false, a1, false, bf, (short)0, acc1, false, false);
        }
    }

    // store: lane<16 -> col=lane rows 0..7 ; lane>=16 -> rows 8..15 (per tile)
    const int   col  = nBase + mr;
    const float bval = bias ? bias[col] : 0.f;
    const int   rowOff = hi ? 8 : 0;
#pragma unroll
    for (int t = 0; t < 2; ++t) {
        v8f acc = t ? acc1 : acc0;
        int base = mBase + t * 16 + rowOff;
#pragma unroll
        for (int v = 0; v < 8; ++v) {
            int mm = base + v;
            float x = acc[v] + bval;
            if (MODE == OUT_RELU_F16) {
                outF16[(size_t)mm * NOUT + col] = (_Float16)(x > 0.f ? x : 0.f);
            } else if (MODE == OUT_H) {
                hbuf[(size_t)mm * 128 + col]   = x;
                outF16[(size_t)mm * 128 + col] = (_Float16)(x > 0.f ? x : 0.f);
            } else { // OUT_ADD_H
                float hh = hbuf[(size_t)mm * 128 + col] + x;
                hbuf[(size_t)mm * 128 + col]   = hh;
                outF16[(size_t)mm * 128 + col] = (_Float16)(hh > 0.f ? hh : 0.f);
            }
        }
    }
}

// ---------------- fused wpre (1x1, 128->4) + cosine VQ argmax + gather
__global__ __launch_bounds__(256) void vq_kernel(
    const float* __restrict__ h, const float* __restrict__ wpre,
    const float* __restrict__ bpre, const float* __restrict__ cbg,
    float* __restrict__ out)
{
    __shared__ float cb[4096 * 4];
    for (int e = threadIdx.x; e < 4096; e += 256) {
        float c0 = cbg[e*4+0], c1 = cbg[e*4+1], c2 = cbg[e*4+2], c3 = cbg[e*4+3];
        float inv = 1.f / (sqrtf(c0*c0 + c1*c1 + c2*c2 + c3*c3) + 1e-10f);
        cb[e*4+0] = c0*inv; cb[e*4+1] = c1*inv; cb[e*4+2] = c2*inv; cb[e*4+3] = c3*inv;
    }
    __syncthreads();

    const int mtok = blockIdx.x * 256 + threadIdx.x;
    float z0 = bpre[0], z1 = bpre[1], z2 = bpre[2], z3 = bpre[3];
    const float* hr = h + (size_t)mtok * 128;
    for (int k = 0; k < 128; k += 4) {
        float4 hv = *(const float4*)(hr + k);
        float h0 = hv.x > 0.f ? hv.x : 0.f;
        float h1 = hv.y > 0.f ? hv.y : 0.f;
        float h2 = hv.z > 0.f ? hv.z : 0.f;
        float h3 = hv.w > 0.f ? hv.w : 0.f;
        z0 += h0*wpre[k]     + h1*wpre[k+1]     + h2*wpre[k+2]     + h3*wpre[k+3];
        z1 += h0*wpre[128+k] + h1*wpre[128+k+1] + h2*wpre[128+k+2] + h3*wpre[128+k+3];
        z2 += h0*wpre[256+k] + h1*wpre[256+k+1] + h2*wpre[256+k+2] + h3*wpre[256+k+3];
        z3 += h0*wpre[384+k] + h1*wpre[384+k+1] + h2*wpre[384+k+2] + h3*wpre[384+k+3];
    }
    float best = -3.4e38f; int bi = 0;
    for (int e = 0; e < 4096; ++e) {
        float s = z0*cb[e*4] + z1*cb[e*4+1] + z2*cb[e*4+2] + z3*cb[e*4+3];
        if (s > best) { best = s; bi = e; }
    }
    const int bb = mtok >> 10, hw = mtok & 1023;
#pragma unroll
    for (int c = 0; c < 4; ++c)
        out[(size_t)(((bb * 4 + c) << 10) + hw)] = cbg[bi * 4 + c];
}

extern "C" void kernel_launch(void* const* d_in, const int* in_sizes, int n_in,
                              void* d_out, int out_size, void* d_ws, size_t ws_size,
                              hipStream_t stream) {
    (void)in_sizes; (void)n_in; (void)out_size; (void)ws_size;
    const float* cond = (const float*)d_in[0];
    const float* w1   = (const float*)d_in[1];
    const float* b1   = (const float*)d_in[2];
    const float* w2   = (const float*)d_in[3];
    const float* b2   = (const float*)d_in[4];
    const float* w3   = (const float*)d_in[5];
    const float* b3   = (const float*)d_in[6];
    const float* rw3  = (const float*)d_in[7];
    const float* rw1  = (const float*)d_in[8];
    const float* wpre = (const float*)d_in[9];
    const float* bpre = (const float*)d_in[10];
    const float* cb   = (const float*)d_in[11];
    float* out = (float*)d_out;

    char* ws = (char*)d_ws;
    _Float16* act1  = (_Float16*)(ws + 0);              // 16 MB [32,64,64,64] NHWC
    _Float16* act2  = (_Float16*)(ws + (16u << 20));    //  8 MB [32768,128]
    float*    hbuf  = (float*)   (ws + (24u << 20));    // 16 MB [32768,128] f32
    _Float16* hrelu = (_Float16*)(ws + (40u << 20));    //  8 MB [32768,128]
    _Float16* r1    = (_Float16*)(ws + (48u << 20));    //  2 MB [32768,32]
    _Float16* bw2   = (_Float16*)(ws + (50u << 20));            // 128*1024
    _Float16* bw3   = bw2 + 128 * 1024;                         // 128*1152
    _Float16* br3   = bw3 + 128 * 1152;                         // 2 x 32*1152
    _Float16* br1   = br3 + 2 * 32 * 1152;                      // 2 x 128*32

    // weight pre-transpose (tiny, L2-resident afterwards)
    prep_weights<128, 1024,  64><<<512, 256, 0, stream>>>(w2, bw2);
    prep_weights<128, 1152, 128><<<576, 256, 0, stream>>>(w3, bw3);
    prep_weights< 32, 1152, 128><<<144, 256, 0, stream>>>(rw3,              br3);
    prep_weights< 32, 1152, 128><<<144, 256, 0, stream>>>(rw3 + 32*128*9,   br3 + 32*1152);
    prep_weights<128,   32,  32><<< 16, 256, 0, stream>>>(rw1,              br1);
    prep_weights<128,   32,  32><<< 16, 256, 0, stream>>>(rw1 + 128*32,     br1 + 128*32);

    conv1_kernel<<<32768, 256, 0, stream>>>(cond, w1, b1, act1);

    // conv2: 4x4 s2 p1, 64x64x64 -> 32x32x128, relu
    gemm_conv<64, 4, 4, 2, 1, 64, 64, 128, OUT_RELU_F16>
        <<<1024, 256, 0, stream>>>(act1, bw2, b2, act2, nullptr);

    // conv3: 3x3 s1 p1, 128->128, bias -> h (f32) + relu(h) (f16)
    gemm_conv<128, 3, 3, 1, 1, 32, 32, 128, OUT_H>
        <<<1024, 256, 0, stream>>>(act2, bw3, b3, hrelu, hbuf);

    for (int i = 0; i < 2; ++i) {
        gemm_conv<128, 3, 3, 1, 1, 32, 32, 32, OUT_RELU_F16>
            <<<1024, 64, 0, stream>>>(hrelu, br3 + (size_t)i * 32 * 1152,
                                      nullptr, r1, nullptr);
        gemm_conv<32, 1, 1, 1, 0, 32, 32, 128, OUT_ADD_H>
            <<<1024, 256, 0, stream>>>(r1, br1 + (size_t)i * 128 * 32,
                                       nullptr, hrelu, hbuf);
    }

    vq_kernel<<<128, 256, 0, stream>>>(hbuf, wpre, bpre, cb, out);
}
